// E34DiagonalWhCell_79551384257206
// MI455X (gfx1250) — matrix operand order repaired and checked by reading, not compiled, not run on hardware
//
#include <hip/hip_runtime.h>

typedef __bf16 v16bf __attribute__((ext_vector_type(16)));
typedef __bf16 v8bf  __attribute__((ext_vector_type(8)));
typedef __bf16 v4bf  __attribute__((ext_vector_type(4)));
typedef float  v8f   __attribute__((ext_vector_type(8)));
typedef float  v4f   __attribute__((ext_vector_type(4)));

#define TT      2048
#define BB      16
#define DD      1024
#define KSTEPS  (DD / 32)        // 32 k-steps of K=32 per timestep
#define PITCH   1032             // bf16 elems per LDS row: 2064B, 16B-aligned, bank-friendly
#define REG16   (16 * PITCH)     // one 16-row region (bf16 elems)
#define LDS_ELEMS (4 * REG16)    // 2 buffers x (hi, lo)

// One block = 4 waves, each wave owns 16 output channels (block covers 64 of D=1024).
// Wave-private W fragments (hi+lo bf16) are held in VGPRs for the whole scan.
extern "C" __global__ void __launch_bounds__(128, 1)
e34_diag_scan_wmma(const float* __restrict__ x, const float* __restrict__ h0,
                   const float* __restrict__ Wx, const float* __restrict__ dvec,
                   const float* __restrict__ bvec, float* __restrict__ out,
                   float* __restrict__ hout)
{
    extern __shared__ __bf16 lds[];

    const int  tid  = threadIdx.x;
    const int  wave = tid >> 5;
    const int  lane = tid & 31;
    const int  lh   = lane & 15;
    const bool hiL  = lane >= 16;
    const int  e    = blockIdx.x * 64 + wave * 16 + lh;   // this lane's output channel

    // ---- Load W row e as B-operand fragments (32x16 bf16 layout), hi/lo split ----
    // B layout: lanes 0-15 hold N=lane, K = k0..k0+7, k0+16..k0+23; lanes 16-31 hold N=lane-16, K = k0+8..k0+15, k0+24..k0+31.
    v16bf wHi[KSTEPS], wLo[KSTEPS];
    {
        const float* wrow = Wx + (size_t)e * DD;
        const int    koff = hiL ? 8 : 0;
#pragma unroll
        for (int kk = 0; kk < KSTEPS; ++kk) {
            v16bf h, l;
#pragma unroll
            for (int j = 0; j < 16; ++j) {
                float  w  = wrow[kk * 32 + koff + j + (j >= 8 ? 8 : 0)];
                __bf16 wh = (__bf16)w;
                h[j] = wh;
                l[j] = (__bf16)(w - (float)wh);
            }
            wHi[kk] = h;
            wLo[kk] = l;
        }
    }

    // ---- Per-channel constants and recurrent state (accumulator layout) ----
    float dd = dvec[e];
    dd = fminf(fmaxf(dd, -0.99f), 0.99f);   // spectral clip
    const float bb = bvec[e];

    float hprev[8];                          // C/D layout: VGPR j -> batch row j (+8 for hi half)
#pragma unroll
    for (int j = 0; j < 8; ++j) {
        const int brow = j + (hiL ? 8 : 0);
        float v  = h0[(size_t)brow * DD + e];
        hprev[j] = v;
        hout[(size_t)brow * DD + e] = v;     // h[0] = h0
    }

    // ---- Stage x[t] into LDS as bf16 hi/lo (shared by all 4 waves) ----
    auto stage = [&](int t, int buf) {
        const v4f* src = (const v4f*)(x + (size_t)t * (BB * DD));
        __bf16* dh = lds + buf * (2 * REG16);
        __bf16* dl = dh + REG16;
#pragma unroll
        for (int i = 0; i < (BB * DD / 4) / 128; ++i) {   // 32 float4 per thread
            int v  = tid + i * 128;
            v4f f  = src[v];
            int n  = v * 4;
            int br = n >> 10;                 // batch row
            int k  = n & 1023;                // k offset
            v4bf ph, pl;
#pragma unroll
            for (int c2 = 0; c2 < 4; ++c2) {
                __bf16 fh = (__bf16)f[c2];
                ph[c2] = fh;
                pl[c2] = (__bf16)(f[c2] - (float)fh);
            }
            *(v4bf*)(dh + br * PITCH + k) = ph;   // ds_store_b64
            *(v4bf*)(dl + br * PITCH + k) = pl;
        }
    };

    stage(0, 0);
    __syncthreads();

    const int s = hiL ? 8 : 0;               // A-fragment K sub-offset for hi lane half

#pragma unroll 1
    for (int t = 0; t < TT; ++t) {
        const int cur = t & 1;
        if (t + 1 < TT) stage(t + 1, cur ^ 1);   // global-load latency overlaps the k-loop

        const __bf16* xh = lds + cur * (2 * REG16) + lh * PITCH;  // A row = lane&15
        const __bf16* xl = xh + REG16;

        // Depth-2 software-pipelined A-fragment fetch (rotating register sets) so
        // ds_load latency hides under the WMMAs of the two preceding k-steps.
        v8bf pa0[3], pa1[3], pb0[3], pb1[3];
#pragma unroll
        for (int p = 0; p < 2; ++p) {
            const int k0 = p * 32;
            pa0[p] = *(const v8bf*)(xh + k0 + s);
            pa1[p] = *(const v8bf*)(xh + k0 + 16 + s);
            pb0[p] = *(const v8bf*)(xl + k0 + s);
            pb1[p] = *(const v8bf*)(xl + k0 + 16 + s);
        }

        v8f c0 = {}, c1 = {}, c2 = {};       // one accumulator per hi/lo pass
#pragma unroll
        for (int kk = 0; kk < KSTEPS; ++kk) {
            const int cb = kk % 3;
            if (kk + 2 < KSTEPS) {           // prefetch k-step kk+2 before this step's WMMAs
                const int nb = (kk + 2) % 3;
                const int k0 = (kk + 2) * 32;
                pa0[nb] = *(const v8bf*)(xh + k0 + s);
                pa1[nb] = *(const v8bf*)(xh + k0 + 16 + s);
                pb0[nb] = *(const v8bf*)(xl + k0 + s);
                pb1[nb] = *(const v8bf*)(xl + k0 + 16 + s);
            }
            v16bf aH = __builtin_shufflevector(pa0[cb], pa1[cb],
                                               0,1,2,3,4,5,6,7,8,9,10,11,12,13,14,15);
            v16bf aL = __builtin_shufflevector(pb0[cb], pb1[cb],
                                               0,1,2,3,4,5,6,7,8,9,10,11,12,13,14,15);
            // hi/lo split GEMM: hi*hi + hi*lo + lo*hi
            c0 = __builtin_amdgcn_wmma_f32_16x16x32_bf16(false, aH, false, wHi[kk], (short)0, c0, false, false);
            c1 = __builtin_amdgcn_wmma_f32_16x16x32_bf16(false, aH, false, wLo[kk], (short)0, c1, false, false);
            c2 = __builtin_amdgcn_wmma_f32_16x16x32_bf16(false, aL, false, wHi[kk], (short)0, c2, false, false);
        }

        // h_new = tanh(xw + b + d*h_prev); out = h_new^2 * sigmoid(h_new)
#pragma unroll
        for (int j = 0; j < 8; ++j) {
            const int brow = j + (hiL ? 8 : 0);
            float z  = (c0[j] + c1[j] + c2[j]) + bb + dd * hprev[j];
            float en = __expf(-2.0f * z);
            float th = (1.0f - en) * __builtin_amdgcn_rcpf(1.0f + en);
            hprev[j] = th;
            float sg = __builtin_amdgcn_rcpf(1.0f + __expf(-th));
            // streamed once, never re-read: keep L2 for x/W with non-temporal stores
            __builtin_nontemporal_store(th * th * sg, &out[((size_t)t * BB + brow) * DD + e]);
            __builtin_nontemporal_store(th,           &hout[((size_t)(t + 1) * BB + brow) * DD + e]);
        }
        __syncthreads();   // buf[cur] fully consumed before it is overwritten at t+2
    }
}

extern "C" void kernel_launch(void* const* d_in, const int* in_sizes, int n_in,
                              void* d_out, int out_size, void* d_ws, size_t ws_size,
                              hipStream_t stream) {
    const float* x  = (const float*)d_in[0];
    const float* h0 = (const float*)d_in[1];
    const float* Wx = (const float*)d_in[2];
    const float* dv = (const float*)d_in[3];
    const float* bv = (const float*)d_in[4];

    float* out  = (float*)d_out;
    float* hout = out + (size_t)TT * BB * DD;

    const size_t shmem = (size_t)LDS_ELEMS * sizeof(__bf16);   // 132096 B (< 320 KB WGP LDS)
    // Allow >64KB dynamic LDS (no-op/ignored error if not required on this runtime).
    (void)hipFuncSetAttribute((const void*)e34_diag_scan_wmma,
                              hipFuncAttributeMaxDynamicSharedMemorySize, (int)shmem);

    e34_diag_scan_wmma<<<DD / 64, 128, shmem, stream>>>(x, h0, Wx, dv, bv, out, hout);
}
